// BHW2AttnRNNDecoder_56032143344040
// MI455X (gfx1250) — matrix operand order repaired and checked by reading, not compiled, not run on hardware
//
#include <hip/hip_runtime.h>
#include <math.h>

// Problem dims
#define B_   32
#define SRC_ 128
#define T_   64
#define H_   512
#define V_   32000
#define H2_  (2*H_)   // 1024
#define H3_  (3*H_)   // 1536

typedef __attribute__((ext_vector_type(8)))  float        v8f;
typedef __attribute__((ext_vector_type(16))) __bf16       v16bf;
typedef __attribute__((ext_vector_type(8)))  __bf16       v8bf;
typedef __attribute__((ext_vector_type(4)))  unsigned int u32x4;
typedef __attribute__((ext_vector_type(8)))  int          i32x8;
typedef __attribute__((ext_vector_type(4)))  int          i32x4;

__device__ __forceinline__ unsigned short f2bf(float f) {
    unsigned u = __float_as_uint(f);
    unsigned r = u + 0x7FFFu + ((u >> 16) & 1u);   // round-to-nearest-even
    return (unsigned short)(r >> 16);
}

// ---------------- one-time kernels ----------------

__global__ void __launch_bounds__(256) k_f32_to_bf16(const float* __restrict__ src,
                                                     unsigned short* __restrict__ dst, int n) {
    int i = blockIdx.x * 256 + threadIdx.x;
    if (i < n) dst[i] = f2bf(src[i]);
}

// encp[b,s,j] = ba[j] + sum_k enc[b,s,k] * Wa[j, H + k]      (loop-invariant attention term)
__global__ void __launch_bounds__(256) k_encproj(const float* __restrict__ enc,
                                                 const float* __restrict__ Wa,
                                                 const float* __restrict__ ba,
                                                 float* __restrict__ encp) {
    int i = blockIdx.x * 256 + threadIdx.x;
    if (i >= B_ * SRC_ * H_) return;
    int j  = i & (H_ - 1);
    int bs = i >> 9;
    const float* e = enc + (size_t)bs * H_;
    const float* w = Wa + (size_t)j * H2_ + H_;
    float acc = ba[j];
    for (int k = 0; k < H_; ++k) acc += e[k] * w[k];
    encp[i] = acc;
}

// ---------------- per-step small kernels (wave-per-output, wave32) ----------------

// hwa[b,j] = sum_k h[b,k] * Wa[j,k]
__global__ void __launch_bounds__(256) k_hWa(const float* __restrict__ h,
                                             const float* __restrict__ Wa,
                                             float* __restrict__ hwa) {
    int w = (blockIdx.x * 256 + threadIdx.x) >> 5;
    int lane = threadIdx.x & 31;
    int b = w >> 9, j = w & (H_ - 1);
    const float* hp = h + (size_t)b * H_;
    const float* wp = Wa + (size_t)j * H2_;
    float acc = 0.f;
    for (int k = lane; k < H_; k += 32) acc += hp[k] * wp[k];
    for (int off = 16; off > 0; off >>= 1) acc += __shfl_xor(acc, off, 32);
    if (lane == 0) hwa[(size_t)b * H_ + j] = acc;
}

// One GRU step; rnn_in = [emb[idx[b,t]], ctx[b]]
__global__ void __launch_bounds__(256) k_gru(const float* __restrict__ hprev,
                                             const float* __restrict__ ctx,
                                             const float* __restrict__ emb,
                                             const int* __restrict__ tidx,
                                             const float* __restrict__ W_ih,
                                             const float* __restrict__ W_hh,
                                             const float* __restrict__ b_ih,
                                             const float* __restrict__ b_hh,
                                             float* __restrict__ hnew, int t) {
    int w = (blockIdx.x * 256 + threadIdx.x) >> 5;
    int lane = threadIdx.x & 31;
    int b = w >> 9, j = w & (H_ - 1);
    const float* xrow = emb + (size_t)tidx[b * T_ + t] * H_;
    const float* crow = ctx + (size_t)b * H_;
    const float* wr = W_ih + (size_t)j * H2_;
    const float* wz = W_ih + (size_t)(H_ + j) * H2_;
    const float* wn = W_ih + (size_t)(2 * H_ + j) * H2_;
    float gr = 0.f, gz = 0.f, gn = 0.f;
    for (int k = lane; k < H_; k += 32) { float x = xrow[k]; gr += x * wr[k];       gz += x * wz[k];       gn += x * wn[k]; }
    for (int k = lane; k < H_; k += 32) { float x = crow[k]; gr += x * wr[H_ + k];  gz += x * wz[H_ + k];  gn += x * wn[H_ + k]; }
    const float* ur = W_hh + (size_t)j * H_;
    const float* uz = W_hh + (size_t)(H_ + j) * H_;
    const float* un = W_hh + (size_t)(2 * H_ + j) * H_;
    const float* hrow = hprev + (size_t)b * H_;
    float hr = 0.f, hz = 0.f, hn = 0.f;
    for (int k = lane; k < H_; k += 32) { float x = hrow[k]; hr += x * ur[k]; hz += x * uz[k]; hn += x * un[k]; }
    for (int off = 16; off > 0; off >>= 1) {
        gr += __shfl_xor(gr, off, 32); gz += __shfl_xor(gz, off, 32); gn += __shfl_xor(gn, off, 32);
        hr += __shfl_xor(hr, off, 32); hz += __shfl_xor(hz, off, 32); hn += __shfl_xor(hn, off, 32);
    }
    if (lane == 0) {
        float r = 1.f / (1.f + expf(-(gr + b_ih[j] + hr + b_hh[j])));
        float z = 1.f / (1.f + expf(-(gz + b_ih[H_ + j] + hz + b_hh[H_ + j])));
        float n = tanhf(gn + b_ih[2 * H_ + j] + r * (hn + b_hh[2 * H_ + j]));
        hnew[(size_t)b * H_ + j] = (1.f - z) * n + z * hrow[j];
    }
}

// scores -> softmax -> context, one block per batch row (512 threads)
__global__ void __launch_bounds__(512) k_attn(const float* __restrict__ hwa,
                                              const float* __restrict__ encp,
                                              const float* __restrict__ enc,
                                              const float* __restrict__ vvec,
                                              float* __restrict__ ctx,
                                              float* __restrict__ attw_out, int t) {
    __shared__ float vsh[H_], hsh[H_], part[512], wsh[SRC_], red[SRC_];
    __shared__ float smax, ssum;
    int tid = threadIdx.x, b = blockIdx.x;
    for (int i = tid; i < H_; i += 512) { vsh[i] = vvec[i]; hsh[i] = hwa[(size_t)b * H_ + i]; }
    __syncthreads();
    int s = tid & (SRC_ - 1), q = tid >> 7;
    const float* ep = encp + ((size_t)b * SRC_ + s) * H_;
    float acc = 0.f;
    int hlo = q * 128;
    for (int h = hlo; h < hlo + 128; ++h) acc += tanhf(hsh[h] + ep[h]) * vsh[h];
    part[tid] = acc;
    __syncthreads();
    if (q == 0) { float e = part[s] + part[s + 128] + part[s + 256] + part[s + 384]; wsh[s] = e; red[s] = e; }
    __syncthreads();
    for (int st = 64; st > 0; st >>= 1) { if (tid < st) red[tid] = fmaxf(red[tid], red[tid + st]); __syncthreads(); }
    if (tid == 0) smax = red[0];
    __syncthreads();
    if (q == 0) { float ex = expf(wsh[s] - smax); wsh[s] = ex; red[s] = ex; }
    __syncthreads();
    for (int st = 64; st > 0; st >>= 1) { if (tid < st) red[tid] += red[tid + st]; __syncthreads(); }
    if (tid == 0) ssum = red[0];
    __syncthreads();
    if (q == 0) wsh[s] = wsh[s] / ssum;
    __syncthreads();
    if (q == 0 && attw_out) attw_out[((size_t)b * T_ + t) * SRC_ + s] = wsh[s];
    // context[b, tid] = sum_s w[s] * enc[b,s,tid]    (tid in [0,512) == H_)
    const float* ebase = enc + (size_t)b * SRC_ * H_ + tid;
    float c = 0.f;
    for (int s2 = 0; s2 < SRC_; ++s2) c += wsh[s2] * ebase[(size_t)s2 * H_];
    ctx[(size_t)b * H_ + tid] = c;
}

// pack [h, ctx] (32 x 1024) to bf16 row-major for the WMMA A operand
__global__ void __launch_bounds__(256) k_pack(const float* __restrict__ h,
                                              const float* __restrict__ ctx,
                                              unsigned short* __restrict__ xc) {
    int i = blockIdx.x * 256 + threadIdx.x;           // over B*H2
    int b = i >> 10, c = i & (H2_ - 1);
    float val = (c < H_) ? h[(size_t)b * H_ + c] : ctx[(size_t)b * H_ + (c - H_)];
    xc[i] = f2bf(val);
}

// logits[b, n] = sum_k XC[b,k] * Wh_bf16[n,k] + bh[n]  -> out[b, t, n]  (NT store)
// XC (64KB) is staged once per workgroup into LDS via the Tensor Data Mover; the
// block's 8 waves then read A fragments via ds_load while streaming B (Wh, L2-resident).
__global__ void __launch_bounds__(256) k_logits(const unsigned short* __restrict__ xc_,
                                                const unsigned short* __restrict__ whb_,
                                                const float* __restrict__ bh,
                                                float* __restrict__ out, int t) {
    __shared__ __align__(16) __bf16 sA[B_ * H2_];   // 64 KB of 320 KB/WGP

#if __has_builtin(__builtin_amdgcn_tensor_load_to_lds) && __has_builtin(__builtin_amdgcn_s_wait_tensorcnt)
    if (threadIdx.x < 32) {   // wave 0 issues one 2D TDM: tile 1024 x 32 elems of 2B
        unsigned long long ga = (unsigned long long)(const void*)xc_;
        unsigned ldsoff = (unsigned)(unsigned long long)(uintptr_t)&sA[0];
        // D# group0: count=1 | lds_addr | global_addr[56:0] | type=2
        u32x4 g0 = { 1u, ldsoff, (unsigned)ga,
                     (unsigned)((ga >> 32) & 0x01FFFFFFu) | (2u << 30) };
        // D# group1: data_size=2B; tensor_dim0=1024, tensor_dim1=32;
        //            tile_dim0=1024, tile_dim1=32; tensor_dim0_stride=1024
        i32x8 g1 = { (int)(1u << 16),          // wg_mask=0, data_size=1 (2 bytes)
                     (int)(1024u << 16),       // tensor_dim0[15:0] in [31:16]
                     (int)(32u << 16),         // tensor_dim0 hi=0, tensor_dim1[15:0]
                     (int)(1024u << 16),       // tensor_dim1 hi=0, tile_dim0
                     32,                       // tile_dim1=32, tile_dim2=0
                     1024,                     // tensor_dim0_stride[31:0]
                     0, 0 };                   // stride hi, tensor_dim1_stride
        i32x4 g2 = { 0, 0, 0, 0 };
        i32x4 g3 = { 0, 0, 0, 0 };
#if __clang_major__ >= 23
        i32x8 g4 = { 0, 0, 0, 0, 0, 0, 0, 0 };
        __builtin_amdgcn_tensor_load_to_lds(g0, g1, g2, g3, g4, 0);
#else
        __builtin_amdgcn_tensor_load_to_lds(g0, g1, g2, g3, 0);
#endif
        __builtin_amdgcn_s_wait_tensorcnt(0);
    }
#else
    // Fallback: cooperative global->LDS copy
    {
        v8bf* dst = (v8bf*)sA;
        const v8bf* src = (const v8bf*)xc_;
        for (int i = threadIdx.x; i < (B_ * H2_) / 8; i += 256) dst[i] = src[i];
    }
#endif
    __syncthreads();

    int lane  = threadIdx.x & 31;
    int wid   = (blockIdx.x * 256 + threadIdx.x) >> 5;
    int n0    = wid * 16;                 // vocab tile base, 250 blocks * 8 waves = 2000 tiles exactly
    int mrow  = lane & 15;                // A: row index / B: column index / D: N index
    int khalf = lane >> 4;                // A: K-half select; B: K 0..7,16..23 vs 8..15,24..31; D: M-half select
    v8f c0 = {}; v8f c1 = {};
    const __bf16* wh  = (const __bf16*)whb_;
    const __bf16* a0p = sA + (size_t)mrow * H2_ + khalf * 8;          // rows 0..15 (LDS)
    const __bf16* a1p = sA + (size_t)(mrow + 16) * H2_ + khalf * 8;   // rows 16..31 (LDS)
    const __bf16* bp  = wh + (size_t)(n0 + mrow) * H2_ + khalf * 8;   // Wh row = B column
    for (int kk = 0; kk < H2_; kk += 32) {
        v8bf a0lo = *(const v8bf*)(a0p + kk);
        v8bf a0hi = *(const v8bf*)(a0p + kk + 16);
        v8bf a1lo = *(const v8bf*)(a1p + kk);
        v8bf a1hi = *(const v8bf*)(a1p + kk + 16);
        v16bf a0, a1;
#pragma unroll
        for (int i = 0; i < 8; ++i) { a0[i] = a0lo[i]; a0[i + 8] = a0hi[i];
                                      a1[i] = a1lo[i]; a1[i + 8] = a1hi[i]; }
        v8bf blo = *(const v8bf*)(bp + kk);
        v8bf bhi = *(const v8bf*)(bp + kk + 16);
        v16bf bfrag;
#pragma unroll
        for (int i = 0; i < 8; ++i) { bfrag[i] = blo[i]; bfrag[i + 8] = bhi[i]; }
        c0 = __builtin_amdgcn_wmma_f32_16x16x32_bf16(false, a0, false, bfrag, (short)0, c0, false, false);
        c1 = __builtin_amdgcn_wmma_f32_16x16x32_bf16(false, a1, false, bfrag, (short)0, c1, false, false);
    }
    float bias = bh[n0 + mrow];
    size_t col = (size_t)(n0 + mrow);
#pragma unroll
    for (int r = 0; r < 8; ++r) {
        int brow = r + khalf * 8;
        __builtin_nontemporal_store(c0[r] + bias, &out[((size_t)brow * T_ + t) * V_ + col]);
        __builtin_nontemporal_store(c1[r] + bias, &out[((size_t)(brow + 16) * T_ + t) * V_ + col]);
    }
}

__global__ void __launch_bounds__(256) k_copy(const float* __restrict__ src,
                                              float* __restrict__ dst, int n) {
    int i = blockIdx.x * 256 + threadIdx.x;
    if (i < n) dst[i] = src[i];
}

// ---------------- host driver ----------------

extern "C" void kernel_launch(void* const* d_in, const int* in_sizes, int n_in,
                              void* d_out, int out_size, void* d_ws, size_t ws_size,
                              hipStream_t stream) {
    const float* enc  = (const float*)d_in[0];   // (B,SRC,H)
    const float* h0   = (const float*)d_in[1];   // (1,B,H)
    const int*   tidx = (const int*)  d_in[2];   // (B,T)
    const float* emb  = (const float*)d_in[3];   // (V,H)
    const float* Wa   = (const float*)d_in[4];   // (H,2H)
    const float* ba   = (const float*)d_in[5];   // (H,)
    const float* vvec = (const float*)d_in[6];   // (1,H)
    const float* W_ih = (const float*)d_in[7];   // (3H,2H)
    const float* W_hh = (const float*)d_in[8];   // (3H,H)
    const float* b_ih = (const float*)d_in[9];
    const float* b_hh = (const float*)d_in[10];
    const float* Wh   = (const float*)d_in[11];  // (V,2H)
    const float* bh   = (const float*)d_in[12];  // (V,)
    (void)in_sizes; (void)n_in; (void)out_size; (void)ws_size;

    float* out_logits = (float*)d_out;                       // (B,T,V)
    float* out_hfin   = out_logits + (size_t)B_ * T_ * V_;   // (1,B,H)
    float* out_attw   = out_hfin + (size_t)B_ * H_;          // (B,T,SRC)

    char* ws = (char*)d_ws;
    size_t off = 0;
    auto alloc = [&](size_t bytes) -> char* {
        char* p = ws + off; off += (bytes + 255) & ~(size_t)255; return p;
    };
    unsigned short* whb  = (unsigned short*)alloc((size_t)V_ * H2_ * 2);   // 65.5 MB, L2-resident
    float*          encp = (float*)         alloc((size_t)B_ * SRC_ * H_ * 4);
    float*          hA   = (float*)         alloc((size_t)B_ * H_ * 4);
    float*          hB   = (float*)         alloc((size_t)B_ * H_ * 4);
    float*          ctx  = (float*)         alloc((size_t)B_ * H_ * 4);
    float*          hwa  = (float*)         alloc((size_t)B_ * H_ * 4);
    unsigned short* xc   = (unsigned short*)alloc((size_t)B_ * H2_ * 2);

    // one-time: Wh -> bf16 (fits MI455X 192MB L2), loop-invariant attention projection
    { int n = V_ * H2_;          k_f32_to_bf16<<<(n + 255) / 256, 256, 0, stream>>>(Wh, whb, n); }
    { int n = B_ * SRC_ * H_;    k_encproj    <<<(n + 255) / 256, 256, 0, stream>>>(enc, Wa, ba, encp); }

    // context0 = attention(h0)
    k_hWa <<<(B_ * H_) / 8, 256, 0, stream>>>(h0, Wa, hwa);
    k_attn<<<B_, 512, 0, stream>>>(hwa, encp, enc, vvec, ctx, (float*)nullptr, 0);

    const float* hprev = h0;
    float* hbufs[2] = { hA, hB };
    for (int t = 0; t < T_; ++t) {
        float* hnew = hbufs[t & 1];
        k_gru   <<<(B_ * H_) / 8, 256, 0, stream>>>(hprev, ctx, emb, tidx, W_ih, W_hh, b_ih, b_hh, hnew, t);
        k_hWa   <<<(B_ * H_) / 8, 256, 0, stream>>>(hnew, Wa, hwa);
        k_attn  <<<B_, 512, 0, stream>>>(hwa, encp, enc, vvec, ctx, out_attw, t);
        k_pack  <<<(B_ * H2_) / 256, 256, 0, stream>>>(hnew, ctx, xc);
        k_logits<<<(V_ / 16) / 8, 256, 0, stream>>>(xc, whb, bh, out_logits, t);
        hprev = hnew;
    }
    { int n = B_ * H_; k_copy<<<(n + 255) / 256, 256, 0, stream>>>(hprev, out_hfin, n); }
}
